// SymbolicEmbeddingsVQ_75943611728221
// MI455X (gfx1250) — matrix operand order repaired and checked by reading, not compiled, not run on hardware
//
#include <hip/hip_runtime.h>
#include <hip/hip_bf16.h>
#include <stdint.h>

// ---------------------------------------------------------------------------
// SymbolicEmbeddingsVQ forward on MI455X (gfx1250, wave32, WMMA).
// out[token, p, :] = symbols[argmin_k (||s_k||^2 - 2 * <latents[id,p,:], s_k>)]
// Dot products via v_wmma_f32_16x16x32_bf16; symbols staged in LDS as bf16.
// 256 blocks (2 WGs/WGP at 132KB LDS) to fill the chip; tile loop unrolled x2.
// ---------------------------------------------------------------------------

typedef __attribute__((ext_vector_type(16))) __bf16          v16bf;
typedef __attribute__((ext_vector_type(16))) unsigned short  v16s;
typedef __attribute__((ext_vector_type(8)))  float           v8f;
typedef __attribute__((ext_vector_type(8)))  unsigned int    v8u;

#define N_SYMBOLS   1024
#define SYMBOL_DIM  64
#define PATTERN_LEN 8
#define N_TOKENS    (8 * 2048)
#define N_GROUPS    (N_TOKENS / 2)   // 8192 groups of 16 rows (2 tokens x 8 rows)
#define BLOCKS      256
#define TPB         256              // 8 wave32 per workgroup

__device__ __forceinline__ unsigned short f2bf(float f) {
  // fp32 -> bf16 round-to-nearest-even
  unsigned u = __float_as_uint(f);
  unsigned r = 0x7FFFu + ((u >> 16) & 1u);
  return (unsigned short)((u + r) >> 16);
}

__device__ __forceinline__ unsigned ordf(float f) {
  // monotone float -> uint mapping (total order matches float <)
  unsigned u = __float_as_uint(f);
  return (u & 0x80000000u) ? ~u : (u | 0x80000000u);
}

__global__ __launch_bounds__(TPB)
void vq_wmma_kernel(const int*   __restrict__ inputs,   // [N_TOKENS]
                    const float* __restrict__ latents,  // [50000, 8, 64]
                    const float* __restrict__ symbols,  // [1024, 64]
                    float*       __restrict__ out)      // [N_TOKENS, 512]
{
  // bf16 symbols, packed 2-per-uint: 1024 rows x 32 uints = 128 KB
  __shared__ unsigned int lds_sym[N_SYMBOLS * (SYMBOL_DIM / 2)];
  __shared__ float        lds_norm[N_SYMBOLS];          // 4 KB

  const int tid = threadIdx.x;

  // ---- Stage symbols into LDS (bf16) and compute ||s||^2 norms ----
  #pragma unroll
  for (int rr = 0; rr < N_SYMBOLS / TPB; ++rr) {
    int s = tid * (N_SYMBOLS / TPB) + rr;
    const float4* src = (const float4*)(symbols + s * SYMBOL_DIM);
    unsigned int* dst = lds_sym + s * (SYMBOL_DIM / 2);
    float nrm = 0.0f;
    #pragma unroll
    for (int q = 0; q < SYMBOL_DIM / 4; ++q) {
      float4 v = src[q];
      nrm = __builtin_fmaf(v.x, v.x, nrm);
      nrm = __builtin_fmaf(v.y, v.y, nrm);
      nrm = __builtin_fmaf(v.z, v.z, nrm);
      nrm = __builtin_fmaf(v.w, v.w, nrm);
      dst[q * 2 + 0] = (unsigned)f2bf(v.x) | ((unsigned)f2bf(v.y) << 16);
      dst[q * 2 + 1] = (unsigned)f2bf(v.z) | ((unsigned)f2bf(v.w) << 16);
    }
    lds_norm[s] = nrm;
  }
  __syncthreads();

  const int lane  = tid & 31;
  const int wave  = tid >> 5;
  const int lhalf = lane >> 4;       // 0: lanes 0-15, 1: lanes 16-31
  const int col   = lane & 15;       // column N within a 16x16 C tile

  // Grid-stride over 16-row groups; trip count is wave-uniform (EXEC stays all-1s).
  for (int g = blockIdx.x * (TPB / 32) + wave; g < N_GROUPS; g += BLOCKS * (TPB / 32)) {
    // ---- Load A tile: 16 rows x 64 dims, bf16, in WMMA A-layout ----
    // Lane L and L+16 both hold row M = L&15 (different K halves).
    const int row   = col;
    const int token = 2 * g + (row >> 3);
    const int id    = inputs[token];
    const float* abase =
        latents + (size_t)id * (PATTERN_LEN * SYMBOL_DIM) + (row & 7) * SYMBOL_DIM;
    const int dOff = lhalf * 16;     // lanes 0-15: K{0..15}; 16-31: K{16..31}

    v16s a0s, a1s;
    #pragma unroll
    for (int j = 0; j < 8; ++j) {
      a0s[j]     = f2bf(abase[dOff + j]);        // WMMA0 K = dOff..dOff+7
      a0s[j + 8] = f2bf(abase[dOff + 8 + j]);    // WMMA0 K = dOff+8..dOff+15
      a1s[j]     = f2bf(abase[dOff + 32 + j]);   // WMMA1 (d = 32 + K)
      a1s[j + 8] = f2bf(abase[dOff + 40 + j]);
    }
    const v16bf A0 = __builtin_bit_cast(v16bf, a0s);
    const v16bf A1 = __builtin_bit_cast(v16bf, a1s);

    float bestd[8];
    int   bests[8];
    #pragma unroll
    for (int i = 0; i < 8; ++i) { bestd[i] = 3.4e38f; bests[i] = 0; }

    const int boff = lhalf * 8;      // B: lanes 0-15 hold K 0..15, lanes 16-31 K 16..31
    const unsigned int* bp0 = lds_sym + col * (SYMBOL_DIM / 2) + boff;

    // ---- Sweep all 1024 symbols in 64 tiles of 16 columns ----
    #pragma unroll 2
    for (int t = 0; t < 64; ++t) {
      const int sym = t * 16 + col;
      const unsigned int* bp = bp0 + t * (16 * (SYMBOL_DIM / 2));
      v8u b0u = *(const v8u*)(bp);               // d  0..31 half for this lane
      v8u b1u = *(const v8u*)(bp + 16);          // d 32..63 half
      const float nrm = lds_norm[sym];

      v8f c = {};
      c = __builtin_amdgcn_wmma_f32_16x16x32_bf16(
              false, A0, false, __builtin_bit_cast(v16bf, b0u), (short)0, c, false, false);
      c = __builtin_amdgcn_wmma_f32_16x16x32_bf16(
              false, A1, false, __builtin_bit_cast(v16bf, b1u), (short)0, c, false, false);

      #pragma unroll
      for (int i = 0; i < 8; ++i) {
        // dist = ||s||^2 - 2*dot ; strict < keeps the lowest sym on ties
        float dist = __builtin_fmaf(-2.0f, c[i], nrm);
        if (dist < bestd[i]) { bestd[i] = dist; bests[i] = sym; }
      }
    }

    // ---- Cross-lane argmin within each 16-lane half (rows i / i+8) ----
    #pragma unroll
    for (int i = 0; i < 8; ++i) {
      unsigned long long key =
          ((unsigned long long)ordf(bestd[i]) << 10) | (unsigned)bests[i];
      #pragma unroll
      for (int m = 1; m <= 8; m <<= 1) {
        unsigned long long o = __shfl_xor(key, m, 32);
        key = (o < key) ? o : key;
      }
      const int sym  = (int)(key & 1023ull);
      const int orow = i + lhalf * 8;
      // Emit the exact fp32 symbol row; 16 lanes x float4 = 64 floats.
      const float4* sp = (const float4*)(symbols + sym * SYMBOL_DIM);
      float4 v = sp[col];
      ((float4*)(out + (size_t)g * 1024 + orow * 64))[col] = v;
    }
  }
}

extern "C" void kernel_launch(void* const* d_in, const int* in_sizes, int n_in,
                              void* d_out, int out_size, void* d_ws, size_t ws_size,
                              hipStream_t stream) {
  const int*   inputs  = (const int*)d_in[0];
  const float* latents = (const float*)d_in[1];
  const float* symbols = (const float*)d_in[2];
  float*       out     = (float*)d_out;
  (void)in_sizes; (void)n_in; (void)out_size; (void)d_ws; (void)ws_size;
  vq_wmma_kernel<<<dim3(BLOCKS), dim3(TPB), 0, stream>>>(inputs, latents, symbols, out);
}
